// ODERNN_31842887533206
// MI455X (gfx1250) — matrix operand order repaired and checked by reading, not compiled, not run on hardware
//
#include <hip/hip_runtime.h>
#include <hip/hip_bf16.h>

// ---------------------------------------------------------------------------
// ODE-RNN for MI455X (gfx1250). Persistent-scan design:
//   grid = 64 blocks, each block owns a 16-row batch tile and runs the full
//   W=200 step scan. 8 waves per block; wave w owns output columns
//   [32w, 32w+32) => 2 WMMA N-tiles. All GEMMs via v_wmma_f32_16x16x32_f16.
//   Weights are pre-packed into WMMA B-fragment layout so the compiler can
//   keep them resident in high VGPRs (VGPR-MSB addressing) across the scan.
//   A-fragments for a whole GEMM are batch-preloaded from LDS so the WMMA
//   chain drains with staged s_wait_dscnt instead of per-kc full waits.
// ---------------------------------------------------------------------------

typedef __attribute__((ext_vector_type(16))) _Float16 v16h;
typedef __attribute__((ext_vector_type(8)))  float    v8f;
typedef __attribute__((ext_vector_type(4)))  float    v4f;

#define B_TOT 1024
#define W_T   200
#define DX    128
#define HD    256
#define DT_MAX 0.05f
#define MAX_SUB 8

// f16 element offsets inside workspace
#define OFF_W1P   0
#define OFF_W2P   (256*256)
#define OFF_WIHP  (2*256*256)
#define OFF_WHHP  (2*256*256 + 768*128)

// ---------------------------------------------------------------------------
// Weight packer: fp32 [N,K] row-major (torch W, output = h @ W^T) -> f16
// WMMA B-fragments. Block bi = ntile*(K/32)+kc is 512 f16 (1KB); lane L holds
// 8 dwords at [L*16 .. L*16+16) f16. dword v = pair (K = kc*32+koff+2v, +1),
// N = ntile*16 + (L&15), koff = (L<16)?0:16.   (contiguous k per lane half; same order as load_A)
// ---------------------------------------------------------------------------
__global__ void pack_b_kernel(const float* __restrict__ src,
                              _Float16* __restrict__ dst, int N, int K) {
    int tid = blockIdx.x * blockDim.x + threadIdx.x;
    int total = (N / 16) * (K / 32) * 256;   // dwords
    if (tid >= total) return;
    int bi = tid >> 8;
    int within = tid & 255;
    int ln = within >> 3;
    int v  = within & 7;
    int kcn = K / 32;
    int ntile = bi / kcn;
    int kc    = bi % kcn;
    int koff = (ln < 16) ? 0 : 16;
    int n = ntile * 16 + (ln & 15);
    int k0 = kc * 32 + koff + 2 * v;
    dst[2 * tid]     = (_Float16)src[n * K + k0];
    dst[2 * tid + 1] = (_Float16)src[n * K + k0 + 1];
}

// ---------------------------------------------------------------------------
// Fragment loaders
// ---------------------------------------------------------------------------
// A (16x32 f16) from LDS row-major [16, ld]; lane = row (L&15);
// lanes 0-15: K = kb+0..15 ; lanes 16-31: K = kb+16..31
__device__ __forceinline__ v16h load_A(const _Float16* __restrict__ buf,
                                       int kbase, int ld) {
    int lane = threadIdx.x & 31;
    int row  = lane & 15;
    int koff = (lane < 16) ? 0 : 16;
    const _Float16* p = buf + row * ld + kbase + koff;
    union { v16h h; v4f f[2]; } u;
    u.f[0] = *(const v4f*)(p);
    u.f[1] = *(const v4f*)(p + 8);
    return u.h;
}

// B (32x16 f16) from packed global blocks: 32B per lane, contiguous.
__device__ __forceinline__ v16h load_B(const _Float16* __restrict__ pack,
                                       int block) {
    int lane = threadIdx.x & 31;
    const _Float16* p = pack + (size_t)block * 512 + lane * 16;
    union { v16h h; v4f f[2]; } u;
    u.f[0] = *(const v4f*)(p);
    u.f[1] = *(const v4f*)(p + 8);
    return u.h;
}

__device__ __forceinline__ v8f wmma16(v16h a, v16h b, v8f c) {
    return __builtin_amdgcn_wmma_f32_16x16x32_f16(
        false, a, false, b, (short)0, c, false, false);
}

// Accumulate A[16,K] @ Bpack tiles (ntile_base, ntile_base+1) into c0,c1.
// All K/32 A-fragments are issued as one ds_load batch up front; the WMMA
// chain then drains with staged dscnt waits (LDS returns are in-order).
__device__ __forceinline__ void gemm2(const _Float16* __restrict__ A_lds, int K,
                                      const _Float16* __restrict__ Bpack,
                                      int ntile_base, v8f& c0, v8f& c1) {
    const int kcn = K / 32;
    v16h afrag[8];
    #pragma unroll
    for (int kc = 0; kc < kcn; ++kc)
        afrag[kc] = load_A(A_lds, kc * 32, K);
    #pragma unroll
    for (int kc = 0; kc < kcn; ++kc) {
        v16h b0 = load_B(Bpack, ntile_base * kcn + kc);
        c0 = wmma16(afrag[kc], b0, c0);
        v16h b1 = load_B(Bpack, (ntile_base + 1) * kcn + kc);
        c1 = wmma16(afrag[kc], b1, c1);
    }
}

__device__ __forceinline__ float sigmoidf(float x) {
    return 1.0f / (1.0f + __expf(-x));
}

// ---------------------------------------------------------------------------
// Main persistent scan kernel
// ---------------------------------------------------------------------------
__global__ __launch_bounds__(256, 1)
void odernn_scan(const float* __restrict__ x, const float* __restrict__ times,
                 const float* __restrict__ b1, const float* __restrict__ b2,
                 const float* __restrict__ b_ih, const float* __restrict__ b_hh,
                 const _Float16* __restrict__ w1p, const _Float16* __restrict__ w2p,
                 const _Float16* __restrict__ wihp, const _Float16* __restrict__ whhp,
                 float* __restrict__ out) {
    __shared__ __align__(16) float    h32s[16 * HD];   // fp32 master hidden
    __shared__ __align__(16) _Float16 h16s[16 * HD];   // f16 mirror (A input)
    __shared__ __align__(16) _Float16 hins[16 * HD];   // RK4 stage input (f16)
    __shared__ __align__(16) _Float16 acts[16 * HD];   // tanh layer output
    __shared__ __align__(16) _Float16 x16s[16 * DX];   // x_t tile (f16)
    __shared__ float dtsub_s[16];
    __shared__ float nsteps_s[16];

    const int tid  = threadIdx.x;
    const int wid  = tid >> 5;
    const int lane = tid & 31;
    const int m0   = blockIdx.x * 16;        // batch-row base
    const int colA = wid * 32 + (lane & 15); // wave-owned columns
    const int colB = colA + 16;
    const int mo   = (lane < 16) ? 0 : 8;    // D-matrix row offset

    // h0 = 0
    for (int i = tid; i < 16 * HD; i += 256) { h32s[i] = 0.0f; h16s[i] = (_Float16)0.0f; }
    __syncthreads();

    #pragma unroll 1
    for (int t = 0; t < W_T; ++t) {
        // ---- load x_t tile (fp32 -> f16) ----
        for (int i = tid; i < 16 * DX; i += 256) {
            int r = i >> 7, d = i & (DX - 1);
            x16s[i] = (_Float16)x[(size_t)(m0 + r) * W_T * DX + (size_t)t * DX + d];
        }
        // ---- per-row dt_sub / n_steps ----
        if (t > 0 && tid < 16) {
            float tp = times[(size_t)(m0 + tid) * W_T + (t - 1)];
            float tc = times[(size_t)(m0 + tid) * W_T + t];
            float dt = fmaxf(tc - tp, 0.0f);
            float d  = fmaxf(dt, 1e-6f);
            float ns = ceilf(d / DT_MAX);
            ns = fminf(fmaxf(ns, 1.0f), 1000.0f);
            dtsub_s[tid]  = d / ns;
            nsteps_s[tid] = ns;
        }
        __syncthreads();

        int smax = 0;
        if (t > 0) {
            float mx = 1.0f;
            for (int r = 0; r < 16; ++r) mx = fmaxf(mx, nsteps_s[r]);
            smax = (int)mx;
            if (smax > MAX_SUB) smax = MAX_SUB;
        }

        // per-lane row constants for this timestep (uniform across substeps)
        float dts8[8], ns8[8];
        #pragma unroll
        for (int i = 0; i < 8; ++i) {
            dts8[i] = dtsub_s[i + mo];
            ns8[i]  = nsteps_s[i + mo];
        }

        // ---- RK4 substeps (block-uniform count; per-row masked commit) ----
        #pragma unroll 1
        for (int s = 0; s < smax; ++s) {
            v8f ks0 = {}, ks1 = {};   // ksum, wave-resident
            // h_base for wave-owned columns, register-cached for the substep
            float hb0[8], hb1[8];
            #pragma unroll
            for (int i = 0; i < 8; ++i) {
                int m = i + mo;
                hb0[i] = h32s[m * HD + colA];
                hb1[i] = h32s[m * HD + colB];
            }
            #pragma unroll 1
            for (int j = 0; j < 4; ++j) {
                const _Float16* Ain = (j == 0) ? h16s : hins;
                // stage 1: act = tanh(Ain @ w1^T + b1), wave's 32 cols
                v8f c0 = {}, c1 = {};
                gemm2(Ain, HD, w1p, wid * 2, c0, c1);
                {
                    float bb0 = b1[colA], bb1 = b1[colB];
                    #pragma unroll
                    for (int i = 0; i < 8; ++i) {
                        int m = i + mo;
                        acts[m * HD + colA] = (_Float16)tanhf(c0[i] + bb0);
                        acts[m * HD + colB] = (_Float16)tanhf(c1[i] + bb1);
                    }
                }
                __syncthreads();
                // stage 2: k = act @ w2^T + b2; fold into ksum / next input
                v8f d0 = {}, d1 = {};
                gemm2(acts, HD, w2p, wid * 2, d0, d1);
                float bb0 = b2[colA], bb1 = b2[colB];
                if (j < 3) {
                    float cw = (j == 2) ? 1.0f : 0.5f;   // h + cw*dt*k
                    float kw = (j == 0) ? 1.0f : 2.0f;   // ksum weight
                    #pragma unroll
                    for (int i = 0; i < 8; ++i) {
                        int m = i + mo;
                        float k0 = d0[i] + bb0;
                        float k1 = d1[i] + bb1;
                        ks0[i] += kw * k0;
                        ks1[i] += kw * k1;
                        hins[m * HD + colA] = (_Float16)(hb0[i] + cw * dts8[i] * k0);
                        hins[m * HD + colB] = (_Float16)(hb1[i] + cw * dts8[i] * k1);
                    }
                } else {
                    #pragma unroll
                    for (int i = 0; i < 8; ++i) {
                        int m = i + mo;
                        bool active = ns8[i] > (float)s;
                        float k0 = d0[i] + bb0;
                        float k1 = d1[i] + bb1;
                        ks0[i] += k0;
                        ks1[i] += k1;
                        if (active) {
                            float hA = hb0[i] + (dts8[i] * (1.0f / 6.0f)) * ks0[i];
                            float hB = hb1[i] + (dts8[i] * (1.0f / 6.0f)) * ks1[i];
                            h32s[m * HD + colA] = hA; h16s[m * HD + colA] = (_Float16)hA;
                            h32s[m * HD + colB] = hB; h16s[m * HD + colB] = (_Float16)hB;
                        }
                    }
                }
                __syncthreads();
            }
        }

        // ---- GRU cell (gate order r,z,n; r/z accumulate x+h parts fused) ----
        v8f ra0 = {}, ra1 = {}, za0 = {}, za1 = {};
        v8f ia0 = {}, ia1 = {}, na0 = {}, na1 = {};
        gemm2(x16s, DX, wihp,  0 + wid * 2, ra0, ra1);   // ir
        gemm2(x16s, DX, wihp, 16 + wid * 2, za0, za1);   // iz
        gemm2(x16s, DX, wihp, 32 + wid * 2, ia0, ia1);   // in
        gemm2(h16s, HD, whhp,  0 + wid * 2, ra0, ra1);   // + hr
        gemm2(h16s, HD, whhp, 16 + wid * 2, za0, za1);   // + hz
        gemm2(h16s, HD, whhp, 32 + wid * 2, na0, na1);   // hn
        __syncthreads();   // all h16 reads done before h rewrite
        {
            float brA = b_ih[colA]       + b_hh[colA];
            float brB = b_ih[colB]       + b_hh[colB];
            float bzA = b_ih[256 + colA] + b_hh[256 + colA];
            float bzB = b_ih[256 + colB] + b_hh[256 + colB];
            float biA = b_ih[512 + colA], biB = b_ih[512 + colB];
            float bnA = b_hh[512 + colA], bnB = b_hh[512 + colB];
            #pragma unroll
            for (int i = 0; i < 8; ++i) {
                int m = i + mo;
                float rA = sigmoidf(ra0[i] + brA);
                float zA = sigmoidf(za0[i] + bzA);
                float nA = tanhf(ia0[i] + biA + rA * (na0[i] + bnA));
                float hoA = h32s[m * HD + colA];
                float hA = (1.0f - zA) * nA + zA * hoA;
                h32s[m * HD + colA] = hA; h16s[m * HD + colA] = (_Float16)hA;

                float rB = sigmoidf(ra1[i] + brB);
                float zB = sigmoidf(za1[i] + bzB);
                float nB = tanhf(ia1[i] + biB + rB * (na1[i] + bnB));
                float hoB = h32s[m * HD + colB];
                float hB = (1.0f - zB) * nB + zB * hoB;
                h32s[m * HD + colB] = hB; h16s[m * HD + colB] = (_Float16)hB;
            }
        }
        __syncthreads();
    }

    // ---- write final h ----
    for (int i = tid; i < 16 * HD; i += 256) {
        out[(size_t)(m0 + (i >> 8)) * HD + (i & (HD - 1))] = h32s[i];
    }
}

// ---------------------------------------------------------------------------
extern "C" void kernel_launch(void* const* d_in, const int* in_sizes, int n_in,
                              void* d_out, int out_size, void* d_ws, size_t ws_size,
                              hipStream_t stream) {
    const float* x     = (const float*)d_in[0];
    const float* times = (const float*)d_in[1];
    const float* w1    = (const float*)d_in[2];
    const float* b1    = (const float*)d_in[3];
    const float* w2    = (const float*)d_in[4];
    const float* b2    = (const float*)d_in[5];
    const float* w_ih  = (const float*)d_in[6];
    const float* b_ih  = (const float*)d_in[7];
    const float* w_hh  = (const float*)d_in[8];
    const float* b_hh  = (const float*)d_in[9];
    float* out = (float*)d_out;

    _Float16* ws = (_Float16*)d_ws;
    _Float16* w1p  = ws + OFF_W1P;
    _Float16* w2p  = ws + OFF_W2P;
    _Float16* wihp = ws + OFF_WIHP;
    _Float16* whhp = ws + OFF_WHHP;

    // Pack weights into WMMA B-fragment layout (f16).
    {
        int t1 = (256 / 16) * (256 / 32) * 256;       // 32768 dwords
        pack_b_kernel<<<(t1 + 255) / 256, 256, 0, stream>>>(w1, w1p, 256, 256);
        pack_b_kernel<<<(t1 + 255) / 256, 256, 0, stream>>>(w2, w2p, 256, 256);
        int t3 = (768 / 16) * (128 / 32) * 256;       // 49152
        pack_b_kernel<<<(t3 + 255) / 256, 256, 0, stream>>>(w_ih, wihp, 768, 128);
        int t4 = (768 / 16) * (256 / 32) * 256;       // 98304
        pack_b_kernel<<<(t4 + 255) / 256, 256, 0, stream>>>(w_hh, whhp, 768, 256);
    }

    // Persistent scan: 64 blocks x 256 threads (8 waves), 16 batch rows each.
    odernn_scan<<<B_TOT / 16, 256, 0, stream>>>(
        x, times, b1, b2, b_ih, b_hh, w1p, w2p, wihp, whhp, out);
}
